// GATv2Convolution_36627481100821
// MI455X (gfx1250) — compile-verified, matrix operand order checked by reading, not run on hardware
//
#include <hip/hip_runtime.h>
#include <hip/hip_bf16.h>

// ---------------------------------------------------------------------------
// GATv2 two-layer forward for MI455X (gfx1250, wave32, WMMA).
// Layer1 GEMMs via v_wmma_f32_16x16x32_bf16 with pre-packed bf16 B fragments;
// edge softmax via wave-per-edge gather + monotone-uint atomicMax + f32 atomic
// scatter; layer2 (128->2) via VALU wave-per-node dot.
// ---------------------------------------------------------------------------

typedef __attribute__((ext_vector_type(16))) __bf16 v16bf;
typedef __attribute__((ext_vector_type(8)))  float  v8f;

#define NFEAT 165
#define HID   128
#define NCHUNK 6              // ceil(165/32)
#define NEG_SLOPE 0.2f

// Monotone float<->uint mapping so atomicMax(uint) == float max.
__device__ __forceinline__ unsigned fenc(float f) {
  unsigned u = __float_as_uint(f);
  return (u & 0x80000000u) ? ~u : (u | 0x80000000u);
}
__device__ __forceinline__ float fdec(unsigned u) {
  u = (u & 0x80000000u) ? (u & 0x7FFFFFFFu) : ~u;
  return __uint_as_float(u);
}

// ---------------------------------------------------------------------------
// K0: pack W1l|W1r columns into bf16 WMMA B fragments.
// Layout: frag[t][c][lane] = 16 bf16, t in [0,16) (0-7 = W1l tiles, 8-15 = W1r),
// c = K chunk, lane = wave lane. Per-lane element j of the fragment holds
// B[k][n] with n = t%8*16 + (lane&15), k = c*32 + (j/2>>2)*16 + (lane>>4)*8
// + (j/2&3)*2 + (j&1)  (ISA 16-bit 16x32 per-lane K packing).
// ---------------------------------------------------------------------------
__global__ void gatv2_packW(const float* __restrict__ Wl, const float* __restrict__ Wr,
                            __bf16* __restrict__ bp) {
  const int idx = blockIdx.x * blockDim.x + threadIdx.x;   // t*NCHUNK*32 + c*32 + lane
  if (idx >= 16 * NCHUNK * 32) return;
  const int lane = idx & 31;
  const int c    = (idx >> 5) % NCHUNK;
  const int t    = idx / (NCHUNK * 32);
  const float* W = (t < 8) ? Wl : Wr;
  const int nt   = (t < 8) ? t : t - 8;
  const int ncol = nt * 16 + (lane & 15);
  const int hi   = lane >> 4;
  __bf16* dst = bp + (size_t)idx * 16;
  #pragma unroll
  for (int j = 0; j < 8; ++j) {
    const int kb = c * 32 + ((j >> 2) << 4) + (hi << 3) + ((j & 3) << 1);
    dst[2 * j]     = (__bf16)((kb     < NFEAT) ? W[(size_t)kb * HID + ncol]       : 0.0f);
    dst[2 * j + 1] = (__bf16)((kb + 1 < NFEAT) ? W[(size_t)(kb + 1) * HID + ncol] : 0.0f);
  }
}

// ---------------------------------------------------------------------------
// K1: xl1 = x @ W1l + b1l ; xr1 = x @ W1r + b1r     (N x 165) @ (165 x 128)
// One wave computes one 16x16 output tile via WMMA bf16, K in chunks of 32.
// grid = (N/16, 4), block = 128 (4 waves). Wave tile id t = by*4 + wave.
// A converted f32->bf16 on the fly (compiler vectorizes row loads to b128);
// B fragments are single 32-byte coalesced vector loads from the pack buffer.
// ---------------------------------------------------------------------------
__global__ void gatv2_gemm1(const float* __restrict__ x,
                            const __bf16* __restrict__ bpack,
                            const float* __restrict__ bl, const float* __restrict__ br,
                            float* __restrict__ xl, float* __restrict__ xr,
                            int n_nodes) {
  const int lane = threadIdx.x & 31;
  const int wave = threadIdx.x >> 5;
  const int t = blockIdx.y * 4 + wave;      // 0..15
  const float* bia; float* out;
  if (t < 8) { bia = bl; out = xl; }
  else       { bia = br; out = xr; }
  const int n0 = (t & 7) * 16;

  const int m0 = blockIdx.x * 16;
  const int mhalf = lane & 15;
  const int hi    = lane >> 4;

  int mrow = m0 + mhalf;
  if (mrow >= n_nodes) mrow = n_nodes - 1;
  const float* xrow = x + (size_t)mrow * NFEAT;
  const v16bf* bfp = (const v16bf*)bpack + (size_t)t * NCHUNK * 32 + lane;

  v8f acc = {};
  #pragma unroll
  for (int c = 0; c < NCHUNK; ++c) {
    const int k0 = c * 32;
    v16bf afrag;
    #pragma unroll
    for (int j = 0; j < 8; ++j) {
      const int kb = k0 + ((j >> 2) << 4) + (hi << 3) + ((j & 3) << 1);
      const float a0 = (kb     < NFEAT) ? xrow[kb]     : 0.0f;
      const float a1 = (kb + 1 < NFEAT) ? xrow[kb + 1] : 0.0f;
      afrag[2 * j]     = (__bf16)a0;
      afrag[2 * j + 1] = (__bf16)a1;
    }
    const v16bf bfrag = bfp[c * 32];
    acc = __builtin_amdgcn_wmma_f32_16x16x32_bf16(
        /*neg_a=*/false, afrag, /*neg_b=*/false, bfrag,
        /*c_mod=*/(short)0, acc, /*reuse_a=*/false, /*reuse_b=*/false);
  }

  // C/D layout: lane 0-15 -> N=lane, VGPR r -> M=r ; lane 16-31 -> M=r+8.
  const int n = n0 + mhalf;
  const float bv = bia[n];
  #pragma unroll
  for (int r = 0; r < 8; ++r) {
    const int m = m0 + r + (hi << 3);
    if (m < n_nodes) out[(size_t)m * HID + n] = acc[r] + bv;
  }
}

// ---------------------------------------------------------------------------
// Init layer-1 node state: acc1[i][d] = bias1[d]; m1 = 0 (== -inf key); denom=0
// ---------------------------------------------------------------------------
__global__ void gatv2_init1(float* __restrict__ acc, const float* __restrict__ bias,
                            unsigned* __restrict__ m, float* __restrict__ denom,
                            int n_nodes) {
  const long long i = (long long)blockIdx.x * blockDim.x + threadIdx.x;
  const long long tot = (long long)n_nodes * HID;
  if (i < tot) acc[i] = bias[(int)(i & (HID - 1))];
  if (i < n_nodes) { m[i] = 0u; denom[i] = 0.0f; }
}

// ---------------------------------------------------------------------------
// K3: wave per edge: score = att . leaky_relu(xl[src] + xr[dst]); seg-max(dst)
// Edges e in [E, E+N) are the implicit self loops (src=dst=e-E).
// ---------------------------------------------------------------------------
__global__ void gatv2_score1(const int* __restrict__ esrc, const int* __restrict__ edst,
                             const float* __restrict__ xl, const float* __restrict__ xr,
                             const float* __restrict__ att,
                             float* __restrict__ score, unsigned* __restrict__ mmax,
                             int n_edges, int n_nodes) {
  const int lane = threadIdx.x & 31;
  const long long gw = ((long long)blockIdx.x * blockDim.x + threadIdx.x) >> 5;
  const long long etot = (long long)n_edges + n_nodes;
  if (gw >= etot) return;
  const int e = (int)gw;
  int s, d;
  if (e < n_edges) { s = esrc[e]; d = edst[e]; } else { s = d = e - n_edges; }

  const float* pl = xl + (size_t)s * HID;
  const float* pr = xr + (size_t)d * HID;
  float acc = 0.0f;
  #pragma unroll
  for (int j = 0; j < 4; ++j) {
    const int k = lane + (j << 5);
    float v = pl[k] + pr[k];
    v = (v > 0.0f) ? v : NEG_SLOPE * v;
    acc += v * att[k];
  }
  #pragma unroll
  for (int off = 16; off; off >>= 1) acc += __shfl_xor(acc, off, 32);
  if (lane == 0) {
    score[e] = acc;
    atomicMax(&mmax[d], fenc(acc));
  }
}

// ---------------------------------------------------------------------------
// K4: a = exp(score - m[dst]); denom[dst] += a   (thread per edge)
// ---------------------------------------------------------------------------
__global__ void gatv2_exp(const int* __restrict__ edst, const float* __restrict__ score,
                          const unsigned* __restrict__ mmax,
                          float* __restrict__ aexp, float* __restrict__ denom,
                          int n_edges, int n_nodes) {
  const int e = blockIdx.x * blockDim.x + threadIdx.x;
  const int etot = n_edges + n_nodes;
  if (e >= etot) return;
  const int d = (e < n_edges) ? edst[e] : (e - n_edges);
  const float a = __expf(score[e] - fdec(mmax[d]));
  aexp[e] = a;
  atomicAdd(&denom[d], a);
}

// ---------------------------------------------------------------------------
// K5: wave per edge: acc[dst][:] += (a/denom[dst]) * xl[src][:]
// ---------------------------------------------------------------------------
__global__ void gatv2_scatter1(const int* __restrict__ esrc, const int* __restrict__ edst,
                               const float* __restrict__ xl,
                               const float* __restrict__ aexp, const float* __restrict__ denom,
                               float* __restrict__ acc, int n_edges, int n_nodes) {
  const int lane = threadIdx.x & 31;
  const long long gw = ((long long)blockIdx.x * blockDim.x + threadIdx.x) >> 5;
  const long long etot = (long long)n_edges + n_nodes;
  if (gw >= etot) return;
  const int e = (int)gw;
  int s, d;
  if (e < n_edges) { s = esrc[e]; d = edst[e]; } else { s = d = e - n_edges; }
  const float w = aexp[e] / denom[d];
  const float* pl = xl + (size_t)s * HID;
  float* pd = acc + (size_t)d * HID;
  #pragma unroll
  for (int j = 0; j < 4; ++j) {
    const int k = lane + (j << 5);
    atomicAdd(&pd[k], w * pl[k]);
  }
}

__global__ void gatv2_relu(float* __restrict__ a, long long n) {
  const long long i = (long long)blockIdx.x * blockDim.x + threadIdx.x;
  if (i < n) a[i] = fmaxf(a[i], 0.0f);
}

// ---------------------------------------------------------------------------
// K7: layer-2 linears (128 -> 2). N=2: tensor cores useless; wave-per-node
// VALU dot + shuffle reduction is the right shape.
// ---------------------------------------------------------------------------
__global__ void gatv2_linear2(const float* __restrict__ h,
                              const float* __restrict__ W2l, const float* __restrict__ b2l,
                              const float* __restrict__ W2r, const float* __restrict__ b2r,
                              float* __restrict__ xl2, float* __restrict__ xr2,
                              int n_nodes) {
  const int lane = threadIdx.x & 31;
  const long long node = ((long long)blockIdx.x * blockDim.x + threadIdx.x) >> 5;
  if (node >= n_nodes) return;
  const float* hr = h + (size_t)node * HID;
  float sl0 = 0.f, sl1 = 0.f, sr0 = 0.f, sr1 = 0.f;
  #pragma unroll
  for (int j = 0; j < 4; ++j) {
    const int k = lane + (j << 5);
    const float v = hr[k];
    sl0 += v * W2l[k * 2 + 0];  sl1 += v * W2l[k * 2 + 1];
    sr0 += v * W2r[k * 2 + 0];  sr1 += v * W2r[k * 2 + 1];
  }
  #pragma unroll
  for (int off = 16; off; off >>= 1) {
    sl0 += __shfl_xor(sl0, off, 32);
    sl1 += __shfl_xor(sl1, off, 32);
    sr0 += __shfl_xor(sr0, off, 32);
    sr1 += __shfl_xor(sr1, off, 32);
  }
  if (lane == 0) {
    xl2[node * 2 + 0] = sl0 + b2l[0];
    xl2[node * 2 + 1] = sl1 + b2l[1];
    xr2[node * 2 + 0] = sr0 + b2r[0];
    xr2[node * 2 + 1] = sr1 + b2r[1];
  }
}

__global__ void gatv2_init2(float* __restrict__ out, const float* __restrict__ bias,
                            unsigned* __restrict__ m, float* __restrict__ denom,
                            int n_nodes) {
  const int i = blockIdx.x * blockDim.x + threadIdx.x;
  if (i < 2 * n_nodes) out[i] = bias[i & 1];
  if (i < n_nodes) { m[i] = 0u; denom[i] = 0.0f; }
}

__global__ void gatv2_score2(const int* __restrict__ esrc, const int* __restrict__ edst,
                             const float* __restrict__ xl2, const float* __restrict__ xr2,
                             const float* __restrict__ att,
                             float* __restrict__ score, unsigned* __restrict__ mmax,
                             int n_edges, int n_nodes) {
  const int e = blockIdx.x * blockDim.x + threadIdx.x;
  const int etot = n_edges + n_nodes;
  if (e >= etot) return;
  int s, d;
  if (e < n_edges) { s = esrc[e]; d = edst[e]; } else { s = d = e - n_edges; }
  float v0 = xl2[s * 2 + 0] + xr2[d * 2 + 0];
  float v1 = xl2[s * 2 + 1] + xr2[d * 2 + 1];
  v0 = (v0 > 0.0f) ? v0 : NEG_SLOPE * v0;
  v1 = (v1 > 0.0f) ? v1 : NEG_SLOPE * v1;
  const float sc = v0 * att[0] + v1 * att[1];
  score[e] = sc;
  atomicMax(&mmax[d], fenc(sc));
}

__global__ void gatv2_scatter2(const int* __restrict__ esrc, const int* __restrict__ edst,
                               const float* __restrict__ xl2,
                               const float* __restrict__ aexp, const float* __restrict__ denom,
                               float* __restrict__ out, int n_edges, int n_nodes) {
  const int e = blockIdx.x * blockDim.x + threadIdx.x;
  const int etot = n_edges + n_nodes;
  if (e >= etot) return;
  int s, d;
  if (e < n_edges) { s = esrc[e]; d = edst[e]; } else { s = d = e - n_edges; }
  const float w = aexp[e] / denom[d];
  atomicAdd(&out[d * 2 + 0], w * xl2[s * 2 + 0]);
  atomicAdd(&out[d * 2 + 1], w * xl2[s * 2 + 1]);
}

// ---------------------------------------------------------------------------
extern "C" void kernel_launch(void* const* d_in, const int* in_sizes, int n_in,
                              void* d_out, int out_size, void* d_ws, size_t ws_size,
                              hipStream_t stream) {
  const float* x    = (const float*)d_in[0];
  const int*   esrc = (const int*)  d_in[1];
  const int*   edst = (const int*)  d_in[2];
  const float* W1l  = (const float*)d_in[3];
  const float* b1l  = (const float*)d_in[4];
  const float* W1r  = (const float*)d_in[5];
  const float* b1r  = (const float*)d_in[6];
  const float* att1 = (const float*)d_in[7];
  const float* bias1= (const float*)d_in[8];
  const float* W2l  = (const float*)d_in[9];
  const float* b2l  = (const float*)d_in[10];
  const float* W2r  = (const float*)d_in[11];
  const float* b2r  = (const float*)d_in[12];
  const float* att2 = (const float*)d_in[13];
  const float* bias2= (const float*)d_in[14];
  float* out = (float*)d_out;

  const int n_nodes = in_sizes[0] / NFEAT;
  const int n_edges = in_sizes[1];
  const int etot = n_edges + n_nodes;

  // Workspace carve (float units).
  float* ws = (float*)d_ws;
  size_t off = 0;
  float*    xl1    = ws + off; off += (size_t)n_nodes * HID;
  float*    xr1    = ws + off; off += (size_t)n_nodes * HID;
  float*    acc1   = ws + off; off += (size_t)n_nodes * HID;   // becomes h after relu
  float*    score  = ws + off; off += (size_t)etot;
  float*    aexp   = ws + off; off += (size_t)etot;
  unsigned* m1     = (unsigned*)(ws + off); off += (size_t)n_nodes;
  float*    denom1 = ws + off; off += (size_t)n_nodes;
  float*    xl2    = ws + off; off += (size_t)n_nodes * 2;
  float*    xr2    = ws + off; off += (size_t)n_nodes * 2;
  unsigned* m2     = (unsigned*)(ws + off); off += (size_t)n_nodes;
  float*    denom2 = ws + off; off += (size_t)n_nodes;
  __bf16*   bpack  = (__bf16*)(ws + off); off += (size_t)(16 * NCHUNK * 32 * 16) / 2;

  // K0: pack W fragments (one-time, 3072 threads).
  gatv2_packW<<<(16 * NCHUNK * 32 + 255) / 256, 256, 0, stream>>>(W1l, W1r, bpack);

  const int mtiles = (n_nodes + 15) / 16;
  gatv2_gemm1<<<dim3(mtiles, 4), 128, 0, stream>>>(x, bpack, b1l, b1r, xl1, xr1, n_nodes);

  const long long tot1 = (long long)n_nodes * HID;
  const int ib = (int)((tot1 + 255) / 256);
  gatv2_init1<<<ib, 256, 0, stream>>>(acc1, bias1, m1, denom1, n_nodes);

  const int wb = (etot + 7) / 8;        // wave-per-edge, 8 waves / block
  const int tb = (etot + 255) / 256;    // thread-per-edge
  gatv2_score1  <<<wb, 256, 0, stream>>>(esrc, edst, xl1, xr1, att1, score, m1, n_edges, n_nodes);
  gatv2_exp     <<<tb, 256, 0, stream>>>(edst, score, m1, aexp, denom1, n_edges, n_nodes);
  gatv2_scatter1<<<wb, 256, 0, stream>>>(esrc, edst, xl1, aexp, denom1, acc1, n_edges, n_nodes);
  gatv2_relu    <<<ib, 256, 0, stream>>>(acc1, tot1);

  const int lb = (int)(((long long)n_nodes * 32 + 255) / 256);
  gatv2_linear2<<<lb, 256, 0, stream>>>(acc1, W2l, b2l, W2r, b2r, xl2, xr2, n_nodes);

  const int i2b = (2 * n_nodes + 255) / 256;
  gatv2_init2   <<<i2b, 256, 0, stream>>>(out, bias2, m2, denom2, n_nodes);
  gatv2_score2  <<<tb, 256, 0, stream>>>(esrc, edst, xl2, xr2, att2, score, m2, n_edges, n_nodes);
  gatv2_exp     <<<tb, 256, 0, stream>>>(edst, score, m2, aexp, denom2, n_edges, n_nodes);
  gatv2_scatter2<<<tb, 256, 0, stream>>>(esrc, edst, xl2, aexp, denom2, out, n_edges, n_nodes);
}
